// NeuroMUSX_E_27401891349154
// MI455X (gfx1250) — matrix operand order repaired and checked by reading, not compiled, not run on hardware
//
#include <hip/hip_runtime.h>
#include <stdint.h>

#define N_NODES 30000
#define N_EDGES 500000
#define E_TOT   (N_EDGES + N_NODES)   // with self loops
#define NGRAPH  64
#define DHID    128
#define NHEADS  16
#define BN_EPS  1e-5f

typedef __bf16 bf16;
typedef __attribute__((ext_vector_type(16))) __bf16 v16bf;
typedef __attribute__((ext_vector_type(8)))  float  v8f;
typedef __attribute__((ext_vector_type(4)))  unsigned int uvec4;

// ---------------- helpers ----------------
__device__ __forceinline__ unsigned f2ord(float f) {
  unsigned u = __float_as_uint(f);
  return (u & 0x80000000u) ? ~u : (u | 0x80000000u);
}
__device__ __forceinline__ float ord2f(unsigned k) {
  unsigned u = (k & 0x80000000u) ? (k & 0x7fffffffu) : ~k;
  return __uint_as_float(u);
}
__device__ __forceinline__ bf16 f2bf(float f) {   // round-to-nearest-even
  union { unsigned u; float f; } in; in.f = f;
  unsigned r = (in.u + 0x7fffu + ((in.u >> 16) & 1u)) >> 16;
  union { unsigned short s; bf16 b; } out; out.s = (unsigned short)r;
  return out.b;
}

// ---------------- utility kernels ----------------
__global__ void k_zero_f32(float* p, long long n) {
  long long i = (long long)blockIdx.x * blockDim.x + threadIdx.x;
  if (i < n) p[i] = 0.f;
}
__global__ void k_zero_u32(unsigned* p, long long n) {
  long long i = (long long)blockIdx.x * blockDim.x + threadIdx.x;
  if (i < n) p[i] = 0u;
}
__global__ void k_f32_to_bf16(const float* __restrict__ s, bf16* __restrict__ d, long long n) {
  long long i = (long long)blockIdx.x * blockDim.x + threadIdx.x;
  if (i < n) d[i] = f2bf(s[i]);
}
// W: K x OUT f32 row-major  ->  Wt: OUT x K bf16 row-major
__global__ void k_w_transpose_bf16(const float* __restrict__ W, bf16* __restrict__ Wt,
                                   int K, int OUT) {
  int t = blockIdx.x * blockDim.x + threadIdx.x;
  if (t >= K * OUT) return;
  int o = t / K, k = t - o * K;
  Wt[(long long)o * K + k] = f2bf(W[(long long)k * OUT + o]);
}

// ---------------- WMMA GEMM: Y[M x OUT] = Xbf[M x 128] @ W + bias ----------------
// Wt is OUT x 128 bf16 (pre-transposed). One wave per 16x16 tile, K loop 4x32.
__global__ __launch_bounds__(256)
void k_wmma_gemm_bias(const bf16* __restrict__ X, const bf16* __restrict__ Wt,
                      const float* __restrict__ bias, float* __restrict__ Y,
                      int M, int OUT) {
  const int K = DHID;
  int wave = (int)(((long long)blockIdx.x * blockDim.x + threadIdx.x) >> 5);
  int lane = threadIdx.x & 31;
  int tilesN = OUT >> 4;
  int tm = (wave / tilesN) << 4;
  int tn = (wave % tilesN) << 4;
  if (tm >= M) return;                    // wave-uniform: EXEC stays all-ones for WMMA
  int half = lane >> 4;                   // which 16-lane half
  int l15  = lane & 15;
  int col  = tn + l15;

  v8f c;
  float bv = bias[col];
#pragma unroll
  for (int r = 0; r < 8; ++r) c[r] = bv;  // C/D layout: same column for all 8 VGPRs

  const bf16* arow = X  + (long long)(tm + l15) * K;
  const bf16* brow = Wt + (long long)col * K;

#pragma unroll
  for (int kb = 0; kb < K; kb += 32) {
    union { v16bf v; uvec4 q[2]; } a, b;
    // A 16x32 bf16: lanes 0-15 hold K {kb..kb+7, kb+16..kb+23}; lanes 16-31 the +8 halves
    a.q[0] = *(const uvec4*)(arow + kb + half * 8);
    a.q[1] = *(const uvec4*)(arow + kb + 16 + half * 8);
    // B 32x16 bf16: lanes 0-15 hold K {kb..kb+7, kb+16..kb+23} of column col; lanes 16-31 the +8 halves
    b.q[0] = *(const uvec4*)(brow + kb + half * 8);
    b.q[1] = *(const uvec4*)(brow + kb + 16 + half * 8);
    c = __builtin_amdgcn_wmma_f32_16x16x32_bf16(false, a.v, false, b.v,
                                                (short)0, c, false, false);
  }

  float* y = Y + (long long)(tm + half * 8) * OUT + col;
#pragma unroll
  for (int r = 0; r < 8; ++r) y[(long long)r * OUT] = c[r];
}

// ---------------- self-loop edge_attr = mean of incoming ----------------
__global__ void k_loop_accum(const float* __restrict__ ea, const int* __restrict__ dst,
                             float* __restrict__ loopat, float* __restrict__ deg) {
  long long e = (long long)blockIdx.x * blockDim.x + threadIdx.x;
  if (e >= N_EDGES) return;
  int d = dst[e];
  atomicAdd(&deg[d], 1.f);
  atomicAdd(&loopat[2 * d],     ea[2 * e]);
  atomicAdd(&loopat[2 * d + 1], ea[2 * e + 1]);
}
__global__ void k_loop_fin(float* __restrict__ loopat, const float* __restrict__ deg) {
  int n = blockIdx.x * blockDim.x + threadIdx.x;
  if (n >= N_NODES) return;
  float dg = fmaxf(deg[n], 1.f);
  loopat[2 * n]     /= dg;
  loopat[2 * n + 1] /= dg;
}

// ---------------- init layer (K = D_IN = 2, scalar is optimal) ----------------
__global__ void k_init_xlxr(const float* __restrict__ x,
                            const float* __restrict__ Wl, const float* __restrict__ bl,
                            const float* __restrict__ Wr, const float* __restrict__ br,
                            float* __restrict__ xl, float* __restrict__ xr) {
  long long t = (long long)blockIdx.x * blockDim.x + threadIdx.x;
  if (t >= (long long)N_NODES * DHID) return;
  int n = (int)(t >> 7), j = (int)(t & 127);
  float x0 = x[2 * n], x1 = x[2 * n + 1];
  xl[t] = x0 * Wl[j] + x1 * Wl[DHID + j] + bl[j];
  xr[t] = x0 * Wr[j] + x1 * Wr[DHID + j] + br[j];
}

// ---------------- edge phase ----------------
__global__ __launch_bounds__(256)
void k_edge_logits(const float* __restrict__ xl, const float* __restrict__ xr,
                   const float* __restrict__ ea, const float* __restrict__ loopat,
                   const int* __restrict__ src, const int* __restrict__ dst,
                   const float* __restrict__ We, const float* __restrict__ att,
                   float* __restrict__ logit, unsigned* __restrict__ mkey, int C) {
  long long t = (long long)blockIdx.x * blockDim.x + threadIdx.x;
  long long e = t >> 4;
  int h = (int)(t & 15);
  if (e >= E_TOT) return;
  int s, d; float e0, e1;
  if (e < N_EDGES) { s = src[e]; d = dst[e]; e0 = ea[2 * e]; e1 = ea[2 * e + 1]; }
  else { s = d = (int)(e - N_EDGES); e0 = loopat[2 * s]; e1 = loopat[2 * s + 1]; }
  int HC = NHEADS * C;
  const float* pl = xl + (long long)s * HC + h * C;
  const float* pr = xr + (long long)d * HC + h * C;
  float acc = 0.f;
  for (int cc = 0; cc < C; ++cc) {
    int j = h * C + cc;
    float ep = e0 * We[j] + e1 * We[HC + j];
    float v = pl[cc] + pr[cc] + ep;
    v = (v < 0.f) ? 0.2f * v : v;        // leaky_relu(., 0.2)
    acc += v * att[j];
  }
  logit[e * NHEADS + h] = acc;
  atomicMax(&mkey[(long long)d * NHEADS + h], f2ord(acc));   // segment max (L2 atomics)
}

__global__ void k_edge_exp(const int* __restrict__ dst, float* __restrict__ logit,
                           const unsigned* __restrict__ mkey, float* __restrict__ denom) {
  long long t = (long long)blockIdx.x * blockDim.x + threadIdx.x;
  long long e = t >> 4;
  int h = (int)(t & 15);
  if (e >= E_TOT) return;
  int d = (e < N_EDGES) ? dst[e] : (int)(e - N_EDGES);
  float m = ord2f(mkey[(long long)d * NHEADS + h]);
  float v = __expf(logit[e * NHEADS + h] - m);
  logit[e * NHEADS + h] = v;
  atomicAdd(&denom[(long long)d * NHEADS + h], v);
}

__global__ void k_edge_alpha(const int* __restrict__ dst, float* __restrict__ logit,
                             const float* __restrict__ denom) {
  long long t = (long long)blockIdx.x * blockDim.x + threadIdx.x;
  long long e = t >> 4;
  int h = (int)(t & 15);
  if (e >= E_TOT) return;
  int d = (e < N_EDGES) ? dst[e] : (int)(e - N_EDGES);
  logit[e * NHEADS + h] /= denom[(long long)d * NHEADS + h];
}

__global__ __launch_bounds__(256)
void k_edge_agg(const float* __restrict__ xl, const int* __restrict__ src,
                const int* __restrict__ dst, const float* __restrict__ alpha,
                float* __restrict__ agg, int clog2) {
  int hclog2 = 4 + clog2;
  int HC = 1 << hclog2;
  long long t = (long long)blockIdx.x * blockDim.x + threadIdx.x;
  long long e = t >> hclog2;
  int j = (int)(t & (HC - 1));
  if (e >= E_TOT) return;
  int s, d;
  if (e < N_EDGES) { s = src[e]; d = dst[e]; } else { s = d = (int)(e - N_EDGES); }
  float a = alpha[e * NHEADS + (j >> clog2)];
  atomicAdd(&agg[(long long)d * HC + j], xl[(long long)s * HC + j] * a);
}

__global__ void k_add_bias128(const float* __restrict__ agg, const float* __restrict__ bias,
                              float* __restrict__ h) {
  long long t = (long long)blockIdx.x * blockDim.x + threadIdx.x;
  if (t >= (long long)N_NODES * DHID) return;
  h[t] = agg[t] + bias[t & 127];
}

__global__ void k_head_mean(const float* __restrict__ agg, const float* __restrict__ bias,
                            float* __restrict__ out) {
  int n = blockIdx.x * blockDim.x + threadIdx.x;
  if (n >= N_NODES) return;
  float s = 0.f;
#pragma unroll
  for (int h = 0; h < NHEADS; ++h) s += agg[(long long)n * NHEADS + h];
  out[n] = s * (1.f / NHEADS) + bias[0];
}

// ---------------- BatchNorm + LeakyReLU(0.01) ----------------
__global__ __launch_bounds__(256)
void k_bn_reduce(const float* __restrict__ h, float* __restrict__ stats) {
  int ch = blockIdx.x, t = threadIdx.x;
  float s = 0.f, s2 = 0.f;
  for (int r = t; r < N_NODES; r += 256) {
    float v = h[(long long)r * DHID + ch];
    s += v; s2 += v * v;
  }
  __shared__ float sa[256], sb[256];
  sa[t] = s; sb[t] = s2; __syncthreads();
  for (int o = 128; o > 0; o >>= 1) {
    if (t < o) { sa[t] += sa[t + o]; sb[t] += sb[t + o]; }
    __syncthreads();
  }
  if (t == 0) {
    float mu = sa[0] / N_NODES;
    stats[ch] = mu;
    stats[DHID + ch] = sb[0] / N_NODES - mu * mu;   // population variance
  }
}
__global__ void k_bn_apply(float* __restrict__ h, const float* __restrict__ stats,
                           const float* __restrict__ gamma, const float* __restrict__ beta) {
  long long t = (long long)blockIdx.x * blockDim.x + threadIdx.x;
  if (t >= (long long)N_NODES * DHID) return;
  int ch = (int)(t & 127);
  float v = (h[t] - stats[ch]) * rsqrtf(stats[DHID + ch] + BN_EPS) * gamma[ch] + beta[ch];
  h[t] = (v < 0.f) ? 0.01f * v : v;
}

// ---------------- masked segment-mean pooling ----------------
__global__ void k_pool_accum(const float* __restrict__ satf, const int* __restrict__ mask,
                             const int* __restrict__ batch, float* __restrict__ pool) {
  int n = blockIdx.x * blockDim.x + threadIdx.x;
  if (n >= N_NODES) return;
  if (mask[n] == 0) {
    int g = batch[n];
    atomicAdd(&pool[g], satf[n]);
    atomicAdd(&pool[NGRAPH + g], 1.f);
  }
}
__global__ void k_pool_fin(const float* __restrict__ pool, float* __restrict__ out) {
  int g = blockIdx.x * blockDim.x + threadIdx.x;
  if (g >= NGRAPH) return;
  out[g] = pool[g] / fmaxf(pool[NGRAPH + g], 1.f);
}

// ---------------- host orchestration ----------------
static inline int gb(long long n, int b = 256) { return (int)((n + b - 1) / b); }

struct GatW {  // pointers into d_in for one GAT param set (pytree sorted-key order)
  const float *We, *Wl, *Wr, *att, *bias, *bl, *br;
};

static void edge_phase(hipStream_t st,
                       const float* xl, const float* xr,
                       const float* ea, const float* loopat,
                       const int* src, const int* dst,
                       const float* We, const float* att,
                       float* elog, unsigned* mkey, float* denom, float* agg,
                       int C, int clog2) {
  const long long NH = (long long)N_NODES * NHEADS;
  const long long EH = (long long)E_TOT * NHEADS;
  const long long EC = (long long)E_TOT << (4 + clog2);
  k_zero_u32<<<gb(NH), 256, 0, st>>>(mkey, NH);
  k_zero_f32<<<gb(NH), 256, 0, st>>>(denom, NH);
  k_zero_f32<<<gb((long long)N_NODES << (4 + clog2)), 256, 0, st>>>(agg, (long long)N_NODES << (4 + clog2));
  k_edge_logits<<<gb(EH), 256, 0, st>>>(xl, xr, ea, loopat, src, dst, We, att, elog, mkey, C);
  k_edge_exp<<<gb(EH), 256, 0, st>>>(dst, elog, mkey, denom);
  k_edge_alpha<<<gb(EH), 256, 0, st>>>(dst, elog, denom);
  k_edge_agg<<<gb(EC), 256, 0, st>>>(xl, src, dst, elog, agg, clog2);
}

extern "C" void kernel_launch(void* const* d_in, const int* in_sizes, int n_in,
                              void* d_out, int out_size, void* d_ws, size_t ws_size,
                              hipStream_t stream) {
  (void)in_sizes; (void)n_in; (void)out_size; (void)ws_size;
  // ---- input map (top-level insertion order; params flattened pytree, sorted keys) ----
  const float* x  = (const float*)d_in[0];
  const float* ea = (const float*)d_in[1];
  int i = 2;
  const float* bn0_beta  = (const float*)d_in[i++];
  const float* bn0_gamma = (const float*)d_in[i++];
  const float* bnh_beta[2];  const float* bnh_gamma[2];
  for (int l = 0; l < 2; ++l) { bnh_beta[l] = (const float*)d_in[i++]; bnh_gamma[l] = (const float*)d_in[i++]; }
  GatW hid[2], initp, musp, satp;
  auto grab = [&](GatW& g) {   // sorted keys: We, Wl, Wr, att, bias, bl, br
    g.We  = (const float*)d_in[i++]; g.Wl = (const float*)d_in[i++];
    g.Wr  = (const float*)d_in[i++]; g.att = (const float*)d_in[i++];
    g.bias = (const float*)d_in[i++]; g.bl = (const float*)d_in[i++];
    g.br  = (const float*)d_in[i++];
  };
  grab(hid[0]); grab(hid[1]); grab(initp); grab(musp); grab(satp);
  const int* edge_index = (const int*)d_in[i++];
  const int* mask  = (const int*)d_in[i++];
  const int* batch = (const int*)d_in[i++];
  const int* src = edge_index;
  const int* dst = edge_index + N_EDGES;
  float* out = (float*)d_out;

  // ---- workspace carve-up ----
  size_t off = 0;
  auto alloc = [&](size_t bytes) -> char* {
    char* p = (char*)d_ws + off;
    off += (bytes + 255) & ~(size_t)255;
    return p;
  };
  float*    h      = (float*)alloc((size_t)N_NODES * DHID * 4);
  float*    xl     = (float*)alloc((size_t)N_NODES * DHID * 4);
  float*    xr     = (float*)alloc((size_t)N_NODES * DHID * 4);
  float*    agg    = (float*)alloc((size_t)N_NODES * DHID * 4);
  float*    elog   = (float*)alloc((size_t)E_TOT * NHEADS * 4);
  unsigned* mkey   = (unsigned*)alloc((size_t)N_NODES * NHEADS * 4);
  float*    denom  = (float*)alloc((size_t)N_NODES * NHEADS * 4);
  float*    loopat = (float*)alloc((size_t)N_NODES * 2 * 4);
  float*    deg    = (float*)alloc((size_t)N_NODES * 4);
  bf16*     hbf    = (bf16*)alloc((size_t)N_NODES * DHID * 2);
  bf16*     wt_hl[2]; bf16* wt_hr[2];
  for (int l = 0; l < 2; ++l) {
    wt_hl[l] = (bf16*)alloc((size_t)DHID * DHID * 2);
    wt_hr[l] = (bf16*)alloc((size_t)DHID * DHID * 2);
  }
  bf16* wt_ml = (bf16*)alloc((size_t)16 * DHID * 2);
  bf16* wt_mr = (bf16*)alloc((size_t)16 * DHID * 2);
  bf16* wt_sl = (bf16*)alloc((size_t)16 * DHID * 2);
  bf16* wt_sr = (bf16*)alloc((size_t)16 * DHID * 2);
  float* stats   = (float*)alloc(2 * DHID * 4);
  float* satfeat = (float*)alloc((size_t)N_NODES * 4);
  float* pool    = (float*)alloc(2 * NGRAPH * 4);

  const long long ND = (long long)N_NODES * DHID;

  // ---- pre-transpose weights to bf16 (OUT x K) ----
  for (int l = 0; l < 2; ++l) {
    k_w_transpose_bf16<<<gb(DHID * DHID), 256, 0, stream>>>(hid[l].Wl, wt_hl[l], DHID, DHID);
    k_w_transpose_bf16<<<gb(DHID * DHID), 256, 0, stream>>>(hid[l].Wr, wt_hr[l], DHID, DHID);
  }
  k_w_transpose_bf16<<<gb(DHID * 16), 256, 0, stream>>>(musp.Wl, wt_ml, DHID, 16);
  k_w_transpose_bf16<<<gb(DHID * 16), 256, 0, stream>>>(musp.Wr, wt_mr, DHID, 16);
  k_w_transpose_bf16<<<gb(DHID * 16), 256, 0, stream>>>(satp.Wl, wt_sl, DHID, 16);
  k_w_transpose_bf16<<<gb(DHID * 16), 256, 0, stream>>>(satp.Wr, wt_sr, DHID, 16);

  // ---- self-loop edge attrs (mean of incoming) ----
  k_zero_f32<<<gb(2LL * N_NODES), 256, 0, stream>>>(loopat, 2LL * N_NODES);
  k_zero_f32<<<gb(N_NODES), 256, 0, stream>>>(deg, N_NODES);
  k_loop_accum<<<gb(N_EDGES), 256, 0, stream>>>(ea, dst, loopat, deg);
  k_loop_fin<<<gb(N_NODES), 256, 0, stream>>>(loopat, deg);

  // ---- init GAT layer (D_IN=2: scalar transforms) ----
  k_init_xlxr<<<gb(ND), 256, 0, stream>>>(x, initp.Wl, initp.bl, initp.Wr, initp.br, xl, xr);
  edge_phase(stream, xl, xr, ea, loopat, src, dst, initp.We, initp.att,
             elog, mkey, denom, agg, 8, 3);
  k_add_bias128<<<gb(ND), 256, 0, stream>>>(agg, initp.bias, h);
  k_bn_reduce<<<DHID, 256, 0, stream>>>(h, stats);
  k_bn_apply<<<gb(ND), 256, 0, stream>>>(h, stats, bn0_gamma, bn0_beta);

  // ---- hidden GAT layers: WMMA bf16 GEMMs for xl/xr ----
  const int gemm_waves128 = (N_NODES / 16) * (DHID / 16);   // 15000 waves
  const int gemm_waves16  = (N_NODES / 16) * 1;             // 1875 waves
  for (int l = 0; l < 2; ++l) {
    k_f32_to_bf16<<<gb(ND), 256, 0, stream>>>(h, hbf, ND);
    k_wmma_gemm_bias<<<gb((long long)gemm_waves128 * 32), 256, 0, stream>>>(
        hbf, wt_hl[l], hid[l].bl, xl, N_NODES, DHID);
    k_wmma_gemm_bias<<<gb((long long)gemm_waves128 * 32), 256, 0, stream>>>(
        hbf, wt_hr[l], hid[l].br, xr, N_NODES, DHID);
    edge_phase(stream, xl, xr, ea, loopat, src, dst, hid[l].We, hid[l].att,
               elog, mkey, denom, agg, 8, 3);
    k_add_bias128<<<gb(ND), 256, 0, stream>>>(agg, hid[l].bias, h);
    k_bn_reduce<<<DHID, 256, 0, stream>>>(h, stats);
    k_bn_apply<<<gb(ND), 256, 0, stream>>>(h, stats, bnh_gamma[l], bnh_beta[l]);
  }

  // ---- mus / sat heads (HEADS=16, C=1, concat=False) ----
  k_f32_to_bf16<<<gb(ND), 256, 0, stream>>>(h, hbf, ND);

  k_wmma_gemm_bias<<<gb((long long)gemm_waves16 * 32), 256, 0, stream>>>(
      hbf, wt_ml, musp.bl, xl, N_NODES, 16);
  k_wmma_gemm_bias<<<gb((long long)gemm_waves16 * 32), 256, 0, stream>>>(
      hbf, wt_mr, musp.br, xr, N_NODES, 16);
  edge_phase(stream, xl, xr, ea, loopat, src, dst, musp.We, musp.att,
             elog, mkey, denom, agg, 1, 0);
  k_head_mean<<<gb(N_NODES), 256, 0, stream>>>(agg, musp.bias, out);           // mus -> out[0..N)

  k_wmma_gemm_bias<<<gb((long long)gemm_waves16 * 32), 256, 0, stream>>>(
      hbf, wt_sl, satp.bl, xl, N_NODES, 16);
  k_wmma_gemm_bias<<<gb((long long)gemm_waves16 * 32), 256, 0, stream>>>(
      hbf, wt_sr, satp.br, xr, N_NODES, 16);
  edge_phase(stream, xl, xr, ea, loopat, src, dst, satp.We, satp.att,
             elog, mkey, denom, agg, 1, 0);
  k_head_mean<<<gb(N_NODES), 256, 0, stream>>>(agg, satp.bias, satfeat);

  // ---- masked segment-mean pooling over graphs ----
  k_zero_f32<<<gb(2 * NGRAPH), 256, 0, stream>>>(pool, 2 * NGRAPH);
  k_pool_accum<<<gb(N_NODES), 256, 0, stream>>>(satfeat, mask, batch, pool);
  k_pool_fin<<<1, NGRAPH, 0, stream>>>(pool, out + N_NODES);                    // sat -> out[N..N+G)
}
